// WindowAttention_37950331027930
// MI455X (gfx1250) — matrix operand order repaired and checked by reading, not compiled, not run on hardware
//
#include <hip/hip_runtime.h>
#include <cstdint>
#include <cstddef>

// ---------------------------------------------------------------------------
// Types for CDNA5 WMMA (wave32): D(16x16,f32) = A(16x32,bf16) x B(32x16,bf16) + C
// ---------------------------------------------------------------------------
typedef __attribute__((ext_vector_type(16))) __bf16 v16bf;
typedef __attribute__((ext_vector_type(8)))  float  v8f;

union Frag {
  v16bf v;
  uint4 u[2];
  unsigned short h[16];
};

// Hardware bf16 conversion (RNE) -- lets clang emit v_cvt*bf16_f32 ops
__device__ __forceinline__ unsigned short f2bf(float f) {
  __bf16 h = (__bf16)f;
  return __builtin_bit_cast(unsigned short, h);
}
__device__ __forceinline__ unsigned f2bf_pk(float lo, float hi) {
  return (unsigned)f2bf(lo) | ((unsigned)f2bf(hi) << 16);
}

__device__ __forceinline__ v8f wmma_bf16(const Frag& a, const Frag& b, v8f c) {
  return __builtin_amdgcn_wmma_f32_16x16x32_bf16(false, a.v, false, b.v,
                                                 (short)0, c, false, false);
}

// gfx1250 async copy global -> LDS (ASYNCcnt-tracked), per ISA 08_async_tensor.
__device__ __forceinline__ void async_copy_b128(unsigned lds_byte_addr, const void* gptr) {
  asm volatile("global_load_async_to_lds_b128 %0, %1, off"
               :: "v"(lds_byte_addr), "v"(gptr) : "memory");
}
__device__ __forceinline__ void wait_async0() {
  asm volatile("s_wait_asynccnt 0" ::: "memory");
}

#define C_DIM 512
#define HW    16384   // 128*128
#define NWIN  2048    // 8 * 16 * 16

// ---------------------------------------------------------------------------
// Kernel 0: pack f32 weights into bf16, pre-swizzled to the 16x32 A-fragment
// layout: dst[(((mat*32+mo)*16+ko)*32+lane)*16 + e]
// Contiguous k order: lanes0-15 hold K{0..15},
// lanes16-31 hold K{16..31}; row m = lane%16.
// ---------------------------------------------------------------------------
__global__ __launch_bounds__(256) void pack_w(const float* __restrict__ wq,
                                              const float* __restrict__ wk,
                                              const float* __restrict__ wv,
                                              const float* __restrict__ wo,
                                              unsigned short* __restrict__ dst) {
  int idx  = blockIdx.x * 256 + threadIdx.x;     // 0 .. 4*512*512-1
  int e    = idx & 15;
  int lane = (idx >> 4) & 31;
  int ko   = (idx >> 9) & 15;
  int mo   = (idx >> 13) & 31;
  int mat  = idx >> 18;
  int half = lane >> 4;
  int vr   = e >> 1, p = e & 1;
  int k    = 16 * half + 2 * vr + p;
  int o    = mo * 16 + (lane & 15);
  int c    = ko * 32 + k;
  const float* src = (mat == 0) ? wq : (mat == 1) ? wk : (mat == 2) ? wv : wo;
  dst[idx] = f2bf(src[o * C_DIM + c]);
}

// ---------------------------------------------------------------------------
// Kernel 1: FUSED Q/K/V projection.  Reads x (f32, 256MB) once per o-tile (4 o-tiles); the
// staged LDS tile feeds 3 weight matrices (24 WMMAs per staged tile).
// Block tile 128(o) x 128(tok); 8 waves of 64x32 per projection.
// Outputs (bf16):
//   Q/K : out[(win*64 + t)*512 + o]   (token-major)
//   V   : out[(win*512 + o)*64 + t]   (c-major / V^T)
// ---------------------------------------------------------------------------
__global__ __launch_bounds__(256) void qkv_fused(const float* __restrict__ x,
                                                 const unsigned short* __restrict__ wpq,
                                                 const unsigned short* __restrict__ wpk,
                                                 const unsigned short* __restrict__ wpv,
                                                 const float* __restrict__ bq,
                                                 const float* __restrict__ bk,
                                                 const float* __restrict__ bv,
                                                 unsigned short* __restrict__ Qo,
                                                 unsigned short* __restrict__ Ko,
                                                 unsigned short* __restrict__ Vo) {
  __shared__ unsigned short sX[128][40];   // [token][k] bf16, 80B rows (16B-aligned)

  const int b    = blockIdx.z;
  const int T0   = blockIdx.x * 128;       // token base within batch b
  const int O0   = blockIdx.y * 128;       // output-channel base
  const int tid  = threadIdx.x;
  const int lane = tid & 31;
  const int wid  = tid >> 5;
  const int ow   = wid >> 2;               // 0..1  -> o sub-tile
  const int tw   = wid & 3;                // 0..3  -> token sub-tile
  const int half = lane >> 4, ln = lane & 15;

  const unsigned short* wps[3] = {wpq, wpk, wpv};

  v8f acc[3][4][2] = {};
  const float* xb = x + (size_t)b * C_DIM * HW;

  for (int kk = 0; kk < 16; ++kk) {        // K loop, 32 channels per step
    // Stage 32(k) x 128(tok) f32 -> bf16 LDS transposed to [tok][k].
    // Each thread: 2 k-rows x 8 tokens -> 8 packed dword LDS stores.
    {
      int kp = tid >> 4;                   // k-pair index 0..15
      int tg = tid & 15;                   // token group of 8
      const float* g0 = xb + (size_t)(kk * 32 + 2 * kp) * HW + T0 + tg * 8;
      const float* g1 = g0 + HW;
      float4 a0 = *(const float4*)(g0);
      float4 a1 = *(const float4*)(g0 + 4);
      float4 c0 = *(const float4*)(g1);
      float4 c1 = *(const float4*)(g1 + 4);
      float lo[8] = {a0.x, a0.y, a0.z, a0.w, a1.x, a1.y, a1.z, a1.w};
      float hi[8] = {c0.x, c0.y, c0.z, c0.w, c1.x, c1.y, c1.z, c1.w};
      #pragma unroll
      for (int j = 0; j < 8; ++j)
        *(unsigned*)&sX[tg * 8 + j][2 * kp] = f2bf_pk(lo[j], hi[j]);
    }
    __syncthreads();

    Frag bf[2];
    #pragma unroll
    for (int no = 0; no < 2; ++no) {       // B fragments from LDS (contiguous 32B)
      const unsigned short* ls = &sX[tw * 32 + no * 16 + ln][16 * half];
      bf[no].u[0] = *(const uint4*)(ls);
      bf[no].u[1] = *(const uint4*)(ls + 8);
    }
    #pragma unroll
    for (int m = 0; m < 3; ++m) {
      #pragma unroll
      for (int mo = 0; mo < 4; ++mo) {     // A fragments: pre-swizzled, contiguous
        Frag a;
        const unsigned short* ws =
            wps[m] + ((size_t)((O0 / 16 + ow * 4 + mo) * 16 + kk) * 32 + lane) * 16;
        a.u[0] = *(const uint4*)(ws);
        a.u[1] = *(const uint4*)(ws + 8);
        #pragma unroll
        for (int no = 0; no < 2; ++no)
          acc[m][mo][no] = wmma_bf16(a, bf[no], acc[m][mo][no]);
      }
    }
    __syncthreads();
  }

  // Epilogue: bias add, f32->bf16, scatter into window layouts
  const float* biases[3] = {bq, bk, bv};
  unsigned short* outs[3] = {Qo, Ko, Vo};
  #pragma unroll
  for (int no = 0; no < 2; ++no) {
    int s  = T0 + tw * 32 + no * 16 + ln;  // spatial token within batch
    int hh = s >> 7, wwp = s & 127;
    int win  = b * 256 + (hh >> 3) * 16 + (wwp >> 3);
    int twin = (hh & 7) * 8 + (wwp & 7);
    #pragma unroll
    for (int m = 0; m < 3; ++m)
      #pragma unroll
      for (int mo = 0; mo < 4; ++mo)
        #pragma unroll
        for (int r = 0; r < 8; ++r) {
          int o = O0 + ow * 64 + mo * 16 + r + 8 * half;  // C/D layout row
          float v = acc[m][mo][no][r] + biases[m][o];
          size_t off = (m == 2) ? ((size_t)win * C_DIM + o) * 64 + twin
                                : ((size_t)win * 64 + twin) * C_DIM + o;
          outs[m][off] = f2bf(v);
        }
  }
}

// ---------------------------------------------------------------------------
// Kernel 2: window attention.  One block per window (64 tokens), 4 waves of
// 16 queries.  K and V^T async-staged to LDS (each window fetched once);
// softmax via wave32 shuffles; P transposed through LDS.
// ---------------------------------------------------------------------------
__global__ __launch_bounds__(128) void win_attn(const unsigned short* __restrict__ Q,
                                                const unsigned short* __restrict__ K,
                                                const unsigned short* __restrict__ Vt,
                                                unsigned short* __restrict__ AO) {
  extern __shared__ unsigned short sm[];
  unsigned short* sK = sm;                     // 64*512 bf16 = 64KB
  unsigned short* sV = sm + 64 * C_DIM;        // 512*64 bf16 = 64KB (V^T)
  unsigned short* sP = sm + 2 * 64 * C_DIM;    // 4*16*64 bf16 = 8KB

  const int win  = blockIdx.x;
  const int tid  = threadIdx.x;
  const int lane = tid & 31;
  const int wid  = tid >> 5;
  const int half = lane >> 4, ln = lane & 15;
  const size_t wbase = (size_t)win * 64 * C_DIM;

  // Async stage of K and V^T windows: 2 * 64KB / (128 thr * 16B) = 64 ops/thr
  {
    const unsigned short* kg = K + wbase;
    const unsigned short* vg = Vt + wbase;     // same element count/stride
    unsigned lK = (unsigned)(size_t)sK;        // low 32 bits = LDS byte address
    unsigned lV = (unsigned)(size_t)sV;
    #pragma unroll 4
    for (int i = 0; i < 32; ++i) {
      int eo = (i * 128 + tid) * 8;            // ushort index, 16B per op
      async_copy_b128(lK + eo * 2, kg + eo);
      async_copy_b128(lV + eo * 2, vg + eo);
    }
  }
  wait_async0();
  __syncthreads();

  const int q0 = wid * 16;

  // ---- S = Q K^T ----
  v8f s[4] = {};
  #pragma unroll 4
  for (int kk = 0; kk < 16; ++kk) {
    Frag a;  // A-frag from Q (contiguous 16 k per lane half)
    const unsigned short* qg = Q + wbase + (size_t)(q0 + ln) * C_DIM + kk * 32 + 16 * half;
    a.u[0] = *(const uint4*)(qg);
    a.u[1] = *(const uint4*)(qg + 8);
    #pragma unroll
    for (int t = 0; t < 4; ++t) {
      Frag bf;  // B-frag: lane = key column, 16 contiguous c from LDS
      const unsigned short* ks = sK + (t * 16 + ln) * C_DIM + kk * 32 + 16 * half;
      bf.u[0] = *(const uint4*)(ks);
      bf.u[1] = *(const uint4*)(ks + 8);
      s[t] = wmma_bf16(a, bf, s[t]);
    }
  }

  // ---- softmax over rows (each row spans 16 lanes x 4 tiles) ----
  const float scale = 0.044194173824159216f;  // 1/sqrt(512)
  unsigned short* pw = sP + wid * 16 * 64;
  #pragma unroll
  for (int r = 0; r < 8; ++r) {
    float m = fmaxf(fmaxf(s[0][r], s[1][r]), fmaxf(s[2][r], s[3][r]));
    #pragma unroll
    for (int d = 1; d < 16; d <<= 1) m = fmaxf(m, __shfl_xor(m, d, 32));
    m *= scale;
    float sum = 0.f;
    #pragma unroll
    for (int t = 0; t < 4; ++t) {
      float p = __expf(s[t][r] * scale - m);
      s[t][r] = p;
      sum += p;
    }
    #pragma unroll
    for (int d = 1; d < 16; d <<= 1) sum += __shfl_xor(sum, d, 32);
    float inv = 1.0f / sum;
    int qm = r + 8 * half;                 // C/D layout: row held by this lane
    #pragma unroll
    for (int t = 0; t < 2; ++t)            // pack neighboring tiles? columns differ
      ;
    #pragma unroll
    for (int t = 0; t < 4; ++t)
      pw[qm * 64 + t * 16 + ln] = f2bf(s[t][r] * inv);
  }
  __syncthreads();

  // ---- O = P V ---- (4 chunks of 128 channels; K-dim = 64 keys)
  for (int cc = 0; cc < 4; ++cc) {
    v8f oa[8] = {};
    #pragma unroll
    for (int ks = 0; ks < 2; ++ks) {
      Frag a;  // A-frag of P from LDS (row = query, contiguous 16 keys per lane half)
      const unsigned short* ps = pw + ln * 64 + ks * 32 + 16 * half;
      a.u[0] = *(const uint4*)(ps);
      a.u[1] = *(const uint4*)(ps + 8);
      #pragma unroll
      for (int t = 0; t < 8; ++t) {
        Frag bf;  // B-frag: lane = output channel, contiguous keys from LDS V^T
        const unsigned short* vs =
            sV + (cc * 128 + t * 16 + ln) * 64 + ks * 32 + 16 * half;
        bf.u[0] = *(const uint4*)(vs);
        bf.u[1] = *(const uint4*)(vs + 8);
        oa[t] = wmma_bf16(a, bf, oa[t]);
      }
    }
    #pragma unroll
    for (int t = 0; t < 8; ++t)
      #pragma unroll
      for (int r = 0; r < 8; ++r) {
        int q = q0 + r + 8 * half;
        int c = cc * 128 + t * 16 + ln;
        AO[wbase + (size_t)q * C_DIM + c] = f2bf(oa[t][r]);
      }
  }
}

// ---------------------------------------------------------------------------
// Kernel 3: output projection.  AO is token-major bf16 so B fragments read
// directly (contiguous) from global; result f32 scattered to (B,C,H,W).
// ---------------------------------------------------------------------------
__global__ __launch_bounds__(256) void out_proj(const unsigned short* __restrict__ AO,
                                                const unsigned short* __restrict__ wp,
                                                const float* __restrict__ bias,
                                                float* __restrict__ out) {
  const int G0   = blockIdx.x * 128;       // global window-token base (0..131071)
  const int O0   = blockIdx.y * 128;
  const int tid  = threadIdx.x;
  const int lane = tid & 31;
  const int wid  = tid >> 5;
  const int ow   = wid >> 2, tw = wid & 3;
  const int half = lane >> 4, ln = lane & 15;

  v8f acc[4][2] = {};
  for (int kk = 0; kk < 16; ++kk) {
    Frag a[4], bf[2];
    #pragma unroll
    for (int mo = 0; mo < 4; ++mo) {
      const unsigned short* ws =
          wp + ((size_t)((O0 / 16 + ow * 4 + mo) * 16 + kk) * 32 + lane) * 16;
      a[mo].u[0] = *(const uint4*)(ws);
      a[mo].u[1] = *(const uint4*)(ws + 8);
    }
    #pragma unroll
    for (int no = 0; no < 2; ++no) {
      int g = G0 + tw * 32 + no * 16 + ln;
      const unsigned short* bs = AO + (size_t)g * C_DIM + kk * 32 + 16 * half;
      __builtin_prefetch(bs + C_DIM, 0, 0);   // stream hint -> global_prefetch
      bf[no].u[0] = *(const uint4*)(bs);
      bf[no].u[1] = *(const uint4*)(bs + 8);
    }
    #pragma unroll
    for (int mo = 0; mo < 4; ++mo)
      #pragma unroll
      for (int no = 0; no < 2; ++no)
        acc[mo][no] = wmma_bf16(a[mo], bf[no], acc[mo][no]);
  }

  #pragma unroll
  for (int no = 0; no < 2; ++no) {
    int g   = G0 + tw * 32 + no * 16 + ln;
    int win = g >> 6, t = g & 63;
    int b   = win >> 8, wr = (win >> 4) & 15, wc = win & 15;
    int hh  = wr * 8 + (t >> 3), ww = wc * 8 + (t & 7);
    #pragma unroll
    for (int mo = 0; mo < 4; ++mo)
      #pragma unroll
      for (int r = 0; r < 8; ++r) {
        int o = O0 + ow * 64 + mo * 16 + r + 8 * half;
        out[(((size_t)b * C_DIM + o) * 128 + hh) * 128 + ww] = acc[mo][no][r] + bias[o];
      }
  }
}

// ---------------------------------------------------------------------------
extern "C" void kernel_launch(void* const* d_in, const int* in_sizes, int n_in,
                              void* d_out, int out_size, void* d_ws, size_t ws_size,
                              hipStream_t stream) {
  (void)in_sizes; (void)n_in; (void)out_size; (void)ws_size;
  const float* x  = (const float*)d_in[0];
  const float* wq = (const float*)d_in[1];
  const float* bq = (const float*)d_in[2];
  const float* wk = (const float*)d_in[3];
  const float* bk = (const float*)d_in[4];
  const float* wv = (const float*)d_in[5];
  const float* bv = (const float*)d_in[6];
  const float* wo = (const float*)d_in[7];
  const float* bo = (const float*)d_in[8];

  unsigned short* wsb = (unsigned short*)d_ws;
  const size_t WSZ = (size_t)C_DIM * C_DIM;          // packed ushorts per matrix
  unsigned short* Wq = wsb;
  unsigned short* Wk = wsb + WSZ;
  unsigned short* Wv = wsb + 2 * WSZ;
  unsigned short* Wo = wsb + 3 * WSZ;
  const size_t SZ = (size_t)NWIN * 64 * C_DIM;       // 64Mi bf16 each
  unsigned short* Qb = wsb + 4 * WSZ;
  unsigned short* Kb = Qb + SZ;
  unsigned short* Vb = Kb + SZ;
  unsigned short* Ab = Vb + SZ;

  pack_w<<<4096, 256, 0, stream>>>(wq, wk, wv, wo, wsb);

  dim3 gp(128, 4, 8);   // token tiles x o-tiles x batch
  qkv_fused<<<gp, 256, 0, stream>>>(x, Wq, Wk, Wv, bq, bk, bv, Qb, Kb, Vb);

  size_t attn_lds = (size_t)(2 * 64 * C_DIM + 4 * 16 * 64) * sizeof(unsigned short); // 136KB
  win_attn<<<NWIN, 128, attn_lds, stream>>>(Qb, Kb, Vb, Ab);

  out_proj<<<dim3(1024, 4), 256, 0, stream>>>(Ab, Wo, bo, (float*)d_out);
}
